// HalfAttentionHead_49478023250185
// MI455X (gfx1250) — matrix operand order repaired and checked by reading, not compiled, not run on hardware
//
#include <hip/hip_runtime.h>
#include <hip/hip_bf16.h>

// ---------------------------------------------------------------------------
// HalfAttentionHead for MI455X (gfx1250, wave32, WMMA).
// B=4, T=4096, C=1024, HS=64. Output: [B, 2048, 64] fp32.
//
// Pipeline:
//   0a) split_x_kernel : x -> xhi/xlo bf16 planes (hi/lo split, ~fp32 GEMM
//                        accuracy via 3-term WMMA products; needed because
//                        the top-k row selection is discrete).
//   0b) split_w_kernel : Wq/Wk/Wv -> transposed WThi/WTlo [3][64][1024] bf16
//                        so WMMA B-fragments are contiguous 32B loads.
//   1) qkv_kernel      : q,k,v = x@W via v_wmma_f32_16x16x32_bf16, pure
//                        load+WMMA inner loop (no LDS, no barriers).
//   2) ksum_kernel     : ksum[b][:] = sum_t k[b][t][:]
//   3) rowsum_kernel   : s[b][t] = q[b][t].ksum[b]  (== wei.sum(-1), since
//                        row sums are over unmasked scores: q.(sum k))
//   4a) rank_kernel    : rank under (value desc, index asc) total order
//   4b) compact_kernel : prefix-sum compaction -> sorted top-2048 indices
//   5) attn_kernel     : gathered causal flash attention; 4 waves per 16-row
//                        tile split over column steps, LDS log-sum-exp merge.
// Workspace required: ~79 MB.
// ---------------------------------------------------------------------------

typedef __attribute__((ext_vector_type(16))) __bf16 v16bf;
typedef __attribute__((ext_vector_type(8)))  __bf16 v8bf;
typedef __attribute__((ext_vector_type(4)))  __bf16 v4bf;
typedef __attribute__((ext_vector_type(8)))  float  v8f;

#define TB  4
#define TT  4096
#define TC  1024
#define THS 64
#define TBT (TB * TT)          // 16384 total rows
#define KSEL 2048

__device__ __forceinline__ v8f wmma_bf16(v16bf a, v16bf b, v8f c) {
  // D(f32 16x16) = A(bf16 16x32) * B(bf16 32x16) + C
  return __builtin_amdgcn_wmma_f32_16x16x32_bf16(
      /*neg_a=*/false, a, /*neg_b=*/false, b,
      /*c_mod=*/(short)0, c, /*reuse_a=*/false, /*reuse_b=*/false);
}

// Split fp32 into hi bf16 (exact truncation) + lo bf16 residual.
__device__ __forceinline__ void split_bf16(float x, __bf16& hi, __bf16& lo) {
  union { float f; unsigned u; } t;
  t.f = x;
  t.u &= 0xFFFF0000u;          // truncate mantissa -> exactly bf16-representable
  hi = (__bf16)t.f;
  lo = (__bf16)(x - t.f);
}

__device__ __forceinline__ v16bf cat8(v8bf a, v8bf b) {
  return __builtin_shufflevector(a, b, 0,1,2,3,4,5,6,7,8,9,10,11,12,13,14,15);
}

// ---------------------------------------------------------------------------
// Kernel 0a: x[16384][1024] f32 -> xhi,xlo bf16 planes (row major).
// grid = TBT*TC/1024 = 16384 blocks, 256 threads, 4 elements/thread.
// ---------------------------------------------------------------------------
__global__ __launch_bounds__(256) void split_x_kernel(
    const float* __restrict__ x, __bf16* __restrict__ xhi,
    __bf16* __restrict__ xlo)
{
  const size_t i0 = ((size_t)blockIdx.x * 256 + threadIdx.x) * 4;
  const float4 v = *(const float4*)(x + i0);
  __bf16 h0, l0, h1, l1, h2, l2, h3, l3;
  split_bf16(v.x, h0, l0);
  split_bf16(v.y, h1, l1);
  split_bf16(v.z, h2, l2);
  split_bf16(v.w, h3, l3);
  v4bf hv, lv;
  hv[0] = h0; hv[1] = h1; hv[2] = h2; hv[3] = h3;
  lv[0] = l0; lv[1] = l1; lv[2] = l2; lv[3] = l3;
  *(v4bf*)(xhi + i0) = hv;
  *(v4bf*)(xlo + i0) = lv;
}

// ---------------------------------------------------------------------------
// Kernel 0b: W[1024][64] x3 -> transposed WThi/WTlo [3][64][1024] bf16.
// grid = 3*64*1024/256 = 768 blocks, 256 threads.
// ---------------------------------------------------------------------------
__global__ __launch_bounds__(256) void split_w_kernel(
    const float* __restrict__ Wq, const float* __restrict__ Wk,
    const float* __restrict__ Wv,
    __bf16* __restrict__ wthi, __bf16* __restrict__ wtlo)
{
  const int gid = blockIdx.x * 256 + threadIdx.x;   // [0, 3*64*1024)
  const int mat = gid >> 16;
  const int rem = gid & 65535;
  const int nn  = rem >> 10;                        // output col [0,64)
  const int cc  = rem & 1023;                       // K index   [0,1024)
  const float* W = (mat == 0) ? Wq : ((mat == 1) ? Wk : Wv);
  __bf16 h, l;
  split_bf16(W[(size_t)cc * THS + nn], h, l);
  wthi[gid] = h;
  wtlo[gid] = l;
}

// ---------------------------------------------------------------------------
// Kernel 1: QKV projection. grid = TBT/16 = 1024 blocks, 128 threads (4
// waves); wave w computes output n-tile [16w,16w+16) for q, k and v of one
// 16-row tile. Inner loop: pure contiguous loads + 9 WMMAs per K-step.
// Fragment layouts (wave32):
//  A (bf16 16x32): lane l (m=l&15, h=l>>4): 16-elem chunk at
//    K = k0 + h*16
//  B (bf16 32x16): lane l (n=l&15, h=l>>4): 16-elem chunk of column n at
//    K = k0 + h*16    (contiguous thanks to the W transpose)
//  C/D (f32 16x16): lane l: c[r] = D[r + 8h][n]
// ---------------------------------------------------------------------------
__global__ __launch_bounds__(128) void qkv_kernel(
    const __bf16* __restrict__ xhi, const __bf16* __restrict__ xlo,
    const __bf16* __restrict__ wthi, const __bf16* __restrict__ wtlo,
    float* __restrict__ qf, float* __restrict__ kf,
    __bf16* __restrict__ qb, __bf16* __restrict__ kb,
    __bf16* __restrict__ vbT)
{
  const int tid  = threadIdx.x;
  const int wv   = tid >> 5;
  const int lane = tid & 31;
  const int m    = lane & 15;          // A row == B col for this lane
  const int h    = lane >> 4;
  const int rt   = blockIdx.x;         // row tile [0, 1024)
  const int nc0  = wv * 16;            // this wave's output column base

  const __bf16* xh = xhi + (size_t)(rt * 16 + m) * TC;
  const __bf16* xl = xlo + (size_t)(rt * 16 + m) * TC;
  const __bf16* bh[3];
  const __bf16* bl[3];
#pragma unroll
  for (int mm = 0; mm < 3; ++mm) {
    const size_t o = ((size_t)mm * THS + nc0 + m) * TC;
    bh[mm] = wthi + o;
    bl[mm] = wtlo + o;
  }

  v8f acc[3] = {};
#pragma unroll 2
  for (int k0 = 0; k0 < TC; k0 += 32) {
    const v16bf ahi = *(const v16bf*)(xh + k0 + h * 16);
    const v16bf alo = *(const v16bf*)(xl + k0 + h * 16);
#pragma unroll
    for (int mm = 0; mm < 3; ++mm) {
      const v16bf bhi = *(const v16bf*)(bh[mm] + k0 + h * 16);
      const v16bf blo = *(const v16bf*)(bl[mm] + k0 + h * 16);
      acc[mm] = wmma_bf16(ahi, bhi, acc[mm]);
      acc[mm] = wmma_bf16(ahi, blo, acc[mm]);
      acc[mm] = wmma_bf16(alo, bhi, acc[mm]);
    }
  }

#pragma unroll
  for (int r = 0; r < 8; ++r) {
    const int row = rt * 16 + r + 8 * h;   // global token row [0, TBT)
    const int col = nc0 + m;
    const float qv = acc[0][r], kv = acc[1][r], vv = acc[2][r];
    qf[(size_t)row * THS + col] = qv;
    kf[(size_t)row * THS + col] = kv;
    qb[(size_t)row * THS + col] = (__bf16)qv;
    kb[(size_t)row * THS + col] = (__bf16)kv;
    const int bb = row >> 12;              // batch
    const int t  = row & (TT - 1);         // token within batch
    vbT[((size_t)bb * THS + col) * TT + t] = (__bf16)vv;
  }
}

// ---------------------------------------------------------------------------
// Kernel 2: ksum[b][d] = sum_t k[b][t][d].  grid = B, 256 threads.
// ---------------------------------------------------------------------------
__global__ __launch_bounds__(256) void ksum_kernel(
    const float* __restrict__ kf, float* __restrict__ ksum)
{
  __shared__ float red[256];
  const int b = blockIdx.x, tid = threadIdx.x;
  const int d = tid & 63, part = tid >> 6;
  float a = 0.f;
  for (int t = part; t < TT; t += 4)
    a += kf[((size_t)b * TT + t) * THS + d];
  red[tid] = a;
  __syncthreads();
  if (tid < 64)
    ksum[b * THS + d] = red[d] + red[d + 64] + red[d + 128] + red[d + 192];
}

// ---------------------------------------------------------------------------
// Kernel 3: s[b][t] = q[b][t] . ksum[b].  grid = TBT/256 = 64, 256 threads.
// ---------------------------------------------------------------------------
__global__ __launch_bounds__(256) void rowsum_kernel(
    const float* __restrict__ qf, const float* __restrict__ ksum,
    float* __restrict__ s)
{
  __shared__ float ks[64];
  const int row = blockIdx.x * 256 + threadIdx.x;   // block-uniform batch
  const int b = row >> 12;
  if (threadIdx.x < 64) ks[threadIdx.x] = ksum[b * THS + threadIdx.x];
  __syncthreads();
  const float* qr = qf + (size_t)row * THS;
  float a = 0.f;
#pragma unroll
  for (int d = 0; d < THS; ++d) a += qr[d] * ks[d];
  s[row] = a;
}

// ---------------------------------------------------------------------------
// Kernel 4a: flag[b][i] = (rank of s[b][i] under (value desc, index asc)
// total order) < KSEL.  Ranks are a permutation -> exactly KSEL selected,
// matching jax.lax.top_k + sort.  grid = B*16 = 64 blocks, 256 threads,
// one element per thread scanning an LDS-resident copy of the batch row.
// ---------------------------------------------------------------------------
__global__ __launch_bounds__(256) void rank_kernel(
    const float* __restrict__ s, int* __restrict__ flag)
{
  __shared__ float sv[TT];
  const int b = blockIdx.x >> 4, seg = blockIdx.x & 15;
  for (int i = threadIdx.x; i < TT; i += 256) sv[i] = s[b * TT + i];
  __syncthreads();
  const int i = seg * 256 + threadIdx.x;
  const float si = sv[i];
  int rank = 0;
  for (int j = 0; j < TT; ++j) {
    const float sj = sv[j];
    rank += ((sj > si) || ((sj == si) && (j < i))) ? 1 : 0;
  }
  flag[b * TT + i] = (rank < KSEL) ? 1 : 0;
}

// ---------------------------------------------------------------------------
// Kernel 4b: compact flagged indices (ascending) via block prefix sum.
// grid = B, 1024 threads.
// ---------------------------------------------------------------------------
__global__ __launch_bounds__(1024) void compact_kernel(
    const int* __restrict__ flag, int* __restrict__ idx)
{
  __shared__ int fl[TT];
  __shared__ int csum[1024];
  const int b = blockIdx.x, tid = threadIdx.x;
  for (int i = tid; i < TT; i += 1024) fl[i] = flag[b * TT + i];
  __syncthreads();
  const int base = tid * 4;
  const int mysum = fl[base] + fl[base + 1] + fl[base + 2] + fl[base + 3];
  csum[tid] = mysum;
  __syncthreads();
  for (int off = 1; off < 1024; off <<= 1) {
    const int v = (tid >= off) ? csum[tid - off] : 0;
    __syncthreads();
    csum[tid] += v;
    __syncthreads();
  }
  int pos = csum[tid] - mysum;            // exclusive prefix
  for (int e = 0; e < 4; ++e)
    if (fl[base + e]) idx[b * KSEL + (pos++)] = base + e;
}

// ---------------------------------------------------------------------------
// Kernel 5: gathered causal flash attention.  One block (4 waves) per
// 16-row tile; wave w handles column steps st = w, w+4, w+8, ... with its
// own online-softmax state; partials merged at the end via LDS
// (log-sum-exp merge).  grid = B*KSEL/16 = 512 blocks, 128 threads.
// ---------------------------------------------------------------------------
__global__ __launch_bounds__(128) void attn_kernel(
    const __bf16* __restrict__ qb, const __bf16* __restrict__ kb,
    const __bf16* __restrict__ vbT, const int* __restrict__ idx,
    float* __restrict__ out)
{
  const int tid  = threadIdx.x;
  const int wv   = tid >> 5;
  const int lane = tid & 31;
  const int n    = lane & 15;
  const int h    = lane >> 4;
  const int gt   = blockIdx.x;            // tile id [0, 512)
  const int b    = gt >> 7;               // 128 tiles per batch
  const int tile = gt & 127;

  __shared__ __align__(32) __bf16 Ps[4][16][32];   // per-wave P staging
  __shared__ float mL[4][16];
  __shared__ float lL[4][16];
  __shared__ float accL[4][16][64];

  const int* idxb = idx + b * KSEL;
  int rowtok[8];
#pragma unroll
  for (int r = 0; r < 8; ++r) rowtok[r] = idxb[tile * 16 + r + 8 * h];
  const int jmax = idxb[tile * 16 + 15];  // max (sorted ascending)

  // Q A-fragments for K-dim 0..31 and 32..63 (per-lane 32B contiguous chunks)
  const __bf16* qrow = qb + ((size_t)(b * TT + idxb[tile * 16 + n])) * THS;
  const v16bf aq0 = *(const v16bf*)(qrow + h * 16);
  const v16bf aq1 = *(const v16bf*)(qrow + 32 + h * 16);

  float mrow[8], lrow[8];
#pragma unroll
  for (int r = 0; r < 8; ++r) { mrow[r] = -__builtin_inff(); lrow[r] = 0.f; }
  v8f acc[4] = {};

  const __bf16* kbb = kb  + (size_t)b * TT * THS;
  const __bf16* vbb = vbT + (size_t)b * THS * TT;
  const float scale = 0.03125f;           // C^-0.5 = 1/32

  const int nsteps = (jmax >> 5) + 1;
  for (int st = wv; st < nsteps; st += 4) {
    const int j0 = st << 5;
    if (st + 4 < nsteps) {                // prefetch this wave's next step
      __builtin_prefetch(kbb + (size_t)(j0 + 128 + n) * THS, 0, 3);
      __builtin_prefetch(vbb + (size_t)n * TT + j0 + 128, 0, 3);
    }

    // --- scores: S[16][32] = Q[16x64] @ K[j0:j0+32]^T ---
    v8f s0 = {}, s1 = {};
    {
      const __bf16* kr0 = kbb + (size_t)(j0 + n) * THS;
      s0 = wmma_bf16(aq0, *(const v16bf*)(kr0 + h * 16), s0);
      s0 = wmma_bf16(aq1, *(const v16bf*)(kr0 + 32 + h * 16), s0);
      const __bf16* kr1 = kbb + (size_t)(j0 + 16 + n) * THS;
      s1 = wmma_bf16(aq0, *(const v16bf*)(kr1 + h * 16), s1);
      s1 = wmma_bf16(aq1, *(const v16bf*)(kr1 + 32 + h * 16), s1);
    }

    // --- mask + scale + online softmax (16-lane shfl reductions) ---
    float p0[8], p1[8], red[8];
#pragma unroll
    for (int r = 0; r < 8; ++r) {
      const int row = rowtok[r];
      const float v0 = (j0 + n      <= row) ? s0[r] * scale : -__builtin_inff();
      const float v1 = (j0 + 16 + n <= row) ? s1[r] * scale : -__builtin_inff();
      p0[r] = v0; p1[r] = v1;
      red[r] = fmaxf(v0, v1);
    }
#pragma unroll
    for (int off = 1; off < 16; off <<= 1) {
#pragma unroll
      for (int r = 0; r < 8; ++r)
        red[r] = fmaxf(red[r], __shfl_xor(red[r], off, 32));
    }
    float alpha[8];
#pragma unroll
    for (int r = 0; r < 8; ++r) {
      const float mn = fmaxf(mrow[r], red[r]);
      alpha[r] = (mrow[r] == -__builtin_inff()) ? 0.f : __expf(mrow[r] - mn);
      mrow[r] = mn;
      p0[r] = __expf(p0[r] - mn);   // masked (-inf) -> 0
      p1[r] = __expf(p1[r] - mn);
      red[r] = p0[r] + p1[r];
    }
#pragma unroll
    for (int off = 1; off < 16; off <<= 1) {
#pragma unroll
      for (int r = 0; r < 8; ++r)
        red[r] += __shfl_xor(red[r], off, 32);
    }
#pragma unroll
    for (int r = 0; r < 8; ++r)
      lrow[r] = lrow[r] * alpha[r] + red[r];
#pragma unroll
    for (int nt = 0; nt < 4; ++nt)
#pragma unroll
      for (int r = 0; r < 8; ++r)
        acc[nt][r] *= alpha[r];

    // --- P: C/D layout -> A layout via per-wave LDS tile ---
#pragma unroll
    for (int r = 0; r < 8; ++r) {
      Ps[wv][r + 8 * h][n]      = (__bf16)p0[r];
      Ps[wv][r + 8 * h][n + 16] = (__bf16)p1[r];
    }
    asm volatile("s_wait_dscnt 0" ::: "memory");  // same-wave LDS RAW fence
    const __bf16* prow = &Ps[wv][n][0];           // A-frag row m == lane&15
    const v16bf ap = *(const v16bf*)(prow + h * 16);

    // --- PV: out[16x64] += P[16x32] @ V[j0:j0+32][0:64] ---
#pragma unroll
    for (int nt = 0; nt < 4; ++nt) {
      const v16bf bv =
          *(const v16bf*)(vbb + ((size_t)(nt * 16 + n)) * TT + j0 + h * 16);
      acc[nt] = wmma_bf16(ap, bv, acc[nt]);
    }
  }

  // --- publish per-wave partials ---
  if (n == 0) {                            // lanes 0 / 16 -> rows 0-7 / 8-15
#pragma unroll
    for (int r = 0; r < 8; ++r) {
      mL[wv][r + 8 * h] = mrow[r];
      lL[wv][r + 8 * h] = lrow[r];
    }
  }
#pragma unroll
  for (int nt = 0; nt < 4; ++nt)
#pragma unroll
    for (int r = 0; r < 8; ++r)
      accL[wv][r + 8 * h][nt * 16 + n] = acc[nt][r];
  __syncthreads();

  // --- log-sum-exp merge of the 4 column-split partials + store ---
  for (int e = tid; e < 16 * 64; e += 128) {
    const int rr = e >> 6, cc = e & 63;
    const float m0 = mL[0][rr], m1 = mL[1][rr], m2 = mL[2][rr], m3 = mL[3][rr];
    const float mx = fmaxf(fmaxf(m0, m1), fmaxf(m2, m3));   // finite: wave 0 ran
    const float e0 = __expf(m0 - mx), e1 = __expf(m1 - mx);
    const float e2 = __expf(m2 - mx), e3 = __expf(m3 - mx);
    const float l = lL[0][rr] * e0 + lL[1][rr] * e1 +
                    lL[2][rr] * e2 + lL[3][rr] * e3;
    const float v = accL[0][rr][cc] * e0 + accL[1][rr][cc] * e1 +
                    accL[2][rr][cc] * e2 + accL[3][rr][cc] * e3;
    out[((size_t)(b * KSEL + tile * 16 + rr)) * THS + cc] = v / l;
  }
}

// ---------------------------------------------------------------------------
extern "C" void kernel_launch(void* const* d_in, const int* in_sizes, int n_in,
                              void* d_out, int out_size, void* d_ws, size_t ws_size,
                              hipStream_t stream)
{
  (void)in_sizes; (void)n_in; (void)out_size; (void)ws_size;
  const float* x  = (const float*)d_in[0];
  const float* Wq = (const float*)d_in[1];
  const float* Wk = (const float*)d_in[2];
  const float* Wv = (const float*)d_in[3];
  float* out = (float*)d_out;

  char* base = (char*)d_ws;
  size_t off = 0;
  auto carve = [&](size_t bytes) -> void* {
    void* r = base + off;
    off = (off + bytes + 255) & ~(size_t)255;
    return r;
  };
  __bf16* xhi  = (__bf16*)carve((size_t)TBT * TC * sizeof(__bf16));    // 32 MB
  __bf16* xlo  = (__bf16*)carve((size_t)TBT * TC * sizeof(__bf16));    // 32 MB
  __bf16* wthi = (__bf16*)carve((size_t)3 * THS * TC * sizeof(__bf16));
  __bf16* wtlo = (__bf16*)carve((size_t)3 * THS * TC * sizeof(__bf16));
  float*  qf   = (float*) carve((size_t)TBT * THS * sizeof(float));    // 4 MB
  float*  kf   = (float*) carve((size_t)TBT * THS * sizeof(float));    // 4 MB
  __bf16* qb   = (__bf16*)carve((size_t)TBT * THS * sizeof(__bf16));   // 2 MB
  __bf16* kb   = (__bf16*)carve((size_t)TBT * THS * sizeof(__bf16));   // 2 MB
  __bf16* vbT  = (__bf16*)carve((size_t)TBT * THS * sizeof(__bf16));   // 2 MB
  float*  ksum = (float*) carve((size_t)TB * THS * sizeof(float));
  float*  s    = (float*) carve((size_t)TBT * sizeof(float));
  int*    flag = (int*)   carve((size_t)TBT * sizeof(int));
  int*    idx  = (int*)   carve((size_t)TB * KSEL * sizeof(int));

  split_x_kernel<<<(TBT * TC) / 1024, 256, 0, stream>>>(x, xhi, xlo);
  split_w_kernel<<<(3 * THS * TC) / 256, 256, 0, stream>>>(Wq, Wk, Wv, wthi, wtlo);
  qkv_kernel    <<<TBT / 16, 128, 0, stream>>>(xhi, xlo, wthi, wtlo,
                                               qf, kf, qb, kb, vbT);
  ksum_kernel   <<<TB, 256, 0, stream>>>(kf, ksum);
  rowsum_kernel <<<TBT / 256, 256, 0, stream>>>(qf, ksum, s);
  rank_kernel   <<<TB * 16, 256, 0, stream>>>(s, flag);
  compact_kernel<<<TB, 1024, 0, stream>>>(flag, idx);
  attn_kernel   <<<TB * KSEL / 16, 128, 0, stream>>>(qb, kb, vbT, idx, out);
}
